// SlotAttention_84473416778392
// MI455X (gfx1250) — matrix operand order repaired and checked by reading, not compiled, not run on hardware
//
#include <hip/hip_runtime.h>

typedef __attribute__((ext_vector_type(16))) __bf16 v16bf;
typedef __attribute__((ext_vector_type(8)))  __bf16 v8bf;
typedef __attribute__((ext_vector_type(4)))  __bf16 v4bf;
typedef __attribute__((ext_vector_type(8)))  float  v8f;

#define DIM    256
#define NSLOTS 7
#define NTOK   4096
#define BATCH  64
#define QSCALE 0.0625f   /* 256^-0.5 */
#define A_EPS  1e-8f
#define LN_EPS 1e-5f
#define VSTR   136       /* padded LDS stride (16B aligned, <=2-way bank conflicts) */
#define NSEG   8         /* token segments per batch in attention phase 1 */

union BfPack { v16bf v; v8bf h[2]; };

// Load a WMMA operand as two 16-byte chunks (A and B alike: contiguous 32B at k0+hi*16, chunks +0 / +8).
__device__ __forceinline__ v16bf ld_pair(const __bf16* p0, const __bf16* p1) {
  BfPack u;
  u.h[0] = *(const v8bf*)p0;
  u.h[1] = *(const v8bf*)p1;
  return u.v;
}

__device__ __forceinline__ v8f wmma_bf16(v16bf a, v16bf b, v8f c) {
  return __builtin_amdgcn_wmma_f32_16x16x32_bf16(false, a, false, b, (short)0, c, false, false);
}

__device__ __forceinline__ float block_sum256(float v, float* scratch) {
  __syncthreads();
#pragma unroll
  for (int off = 16; off > 0; off >>= 1) v += __shfl_xor(v, off, 32);
  const int wave = threadIdx.x >> 5, lane = threadIdx.x & 31;
  if (lane == 0) scratch[wave] = v;
  __syncthreads();
  if (threadIdx.x == 0) {
    float s = 0.f;
    for (int w = 0; w < 8; ++w) s += scratch[w];
    scratch[0] = s;
  }
  __syncthreads();
  return scratch[0];
}

// ---------------- small utility kernels ----------------
__global__ void k_cvt_bf16(const float* __restrict__ s, __bf16* __restrict__ d, int n) {
  int i = blockIdx.x * 256 + threadIdx.x;
  if (i < n) d[i] = (__bf16)s[i];
}

__global__ void k_zero_bf16(__bf16* __restrict__ d, int n) {
  int i = blockIdx.x * 256 + threadIdx.x;
  if (i < n) d[i] = (__bf16)0.f;
}

__global__ void k_slots_init(const float* __restrict__ mean, const float* __restrict__ lstd,
                             const float* __restrict__ noise, float* __restrict__ slots, int n) {
  int i = blockIdx.x * 256 + threadIdx.x;
  if (i < n) {
    int d = i & (DIM - 1);
    slots[i] = mean[d] + __expf(lstd[d]) * noise[i];
  }
}

// ---------------- LN(inputs) + keys/values projection (WMMA) ----------------
// grid = B*N/128 blocks of 256 threads (8 waves). keys: [b][n][d] bf16,
// values stored transposed: [b][d][n] bf16. One 70KB LDS buffer reused three
// ways: LN output (xn) -> keys staging -> transposed values staging, so every
// global store is a coalesced b128.
__global__ void __launch_bounds__(256)
k_ln_kv(const float* __restrict__ inp, const float* __restrict__ g,
        const float* __restrict__ bta, const __bf16* __restrict__ WkB,
        const __bf16* __restrict__ WvB, const float* __restrict__ bk,
        const float* __restrict__ bv, __bf16* __restrict__ keys,
        __bf16* __restrict__ vals_t) {
  __shared__ __align__(16) __bf16 buf[DIM * VSTR];   // 69632 B
  const int t = threadIdx.x;
  const int wave = t >> 5, lane = t & 31, hi = lane >> 4, lo = lane & 15;
  const long long r0 = (long long)blockIdx.x * 128;
  const long long bb = r0 >> 12;            // whole block lies in one batch b
  const int nblk = (int)(r0 & 4095);

  // --- 1) LayerNorm, wave-per-row, lane-strided float4 (coalesced) ---
  {
    float4 ga = ((const float4*)g)[lane],   gb = ((const float4*)g)[lane + 32];
    float4 ba = ((const float4*)bta)[lane], bbv = ((const float4*)bta)[lane + 32];
    for (int rr = 0; rr < 16; ++rr) {
      const int rl = wave * 16 + rr;
      const float* x = inp + (r0 + rl) * DIM;
      float4 xa = ((const float4*)x)[lane];
      float4 xb = ((const float4*)x)[lane + 32];
      float s  = xa.x + xa.y + xa.z + xa.w + xb.x + xb.y + xb.z + xb.w;
      float ss = xa.x * xa.x + xa.y * xa.y + xa.z * xa.z + xa.w * xa.w +
                 xb.x * xb.x + xb.y * xb.y + xb.z * xb.z + xb.w * xb.w;
#pragma unroll
      for (int off = 16; off > 0; off >>= 1) {
        s  += __shfl_xor(s, off, 32);
        ss += __shfl_xor(ss, off, 32);
      }
      float mu = s * (1.f / DIM);
      float var = ss * (1.f / DIM) - mu * mu;
      float rs = rsqrtf(var + LN_EPS);
      __bf16* row = buf + rl * DIM;
      v4bf o0, o1;
      o0[0] = (__bf16)((xa.x - mu) * rs * ga.x + ba.x);
      o0[1] = (__bf16)((xa.y - mu) * rs * ga.y + ba.y);
      o0[2] = (__bf16)((xa.z - mu) * rs * ga.z + ba.z);
      o0[3] = (__bf16)((xa.w - mu) * rs * ga.w + ba.w);
      o1[0] = (__bf16)((xb.x - mu) * rs * gb.x + bbv.x);
      o1[1] = (__bf16)((xb.y - mu) * rs * gb.y + bbv.y);
      o1[2] = (__bf16)((xb.z - mu) * rs * gb.z + bbv.z);
      o1[3] = (__bf16)((xb.w - mu) * rs * gb.w + bbv.w);
      *(v4bf*)(row + 4 * lane)       = o0;
      *(v4bf*)(row + 128 + 4 * lane) = o1;
    }
  }
  __syncthreads();

  // --- 2) load A operands (this wave's 16 LN'd rows) into registers ---
  const int m0 = wave * 16;
  v16bf a[8];
  {
    const __bf16* arow = &buf[(m0 + lo) * DIM];
#pragma unroll
    for (int c = 0; c < 8; ++c) {
      int kA = c * 32 + hi * 16;
      a[c] = ld_pair(arow + kA, arow + kA + 8);
    }
  }
  __syncthreads();   // buf free for reuse now

  // --- 3) keys = xn @ Wk^T  -> stage bf16 in buf[row][j] ---
  for (int jt = 0; jt < 16; ++jt) {
    const int j = jt * 16 + lo;
    v8f acc = {};
#pragma unroll
    for (int c = 0; c < 8; ++c) {
      const __bf16* p = WkB + j * DIM + c * 32 + hi * 16;
      acc = wmma_bf16(a[c], ld_pair(p, p + 8), acc);
    }
    float bj = bk[j];
#pragma unroll
    for (int r = 0; r < 8; ++r)
      buf[(m0 + r + 8 * hi) * DIM + j] = (__bf16)(acc[r] + bj);
  }
  __syncthreads();
  // cooperative coalesced store: 128 rows x 512B, 16B per thread per round
  {
    __bf16* dst = keys + r0 * DIM;
#pragma unroll
    for (int i = 0; i < 16; ++i) {
      int off = (i * 256 + t) * 8;
      *(v8bf*)(dst + off) = *(const v8bf*)(buf + off);
    }
  }
  __syncthreads();

  // --- 4) values = xn @ Wv^T -> stage transposed in buf[j][nloc] (stride VSTR) ---
  for (int jt = 0; jt < 16; ++jt) {
    const int j = jt * 16 + lo;
    v8f acv = {};
#pragma unroll
    for (int c = 0; c < 8; ++c) {
      const __bf16* p = WvB + j * DIM + c * 32 + hi * 16;
      acv = wmma_bf16(a[c], ld_pair(p, p + 8), acv);
    }
    float vj = bv[j];
#pragma unroll
    for (int r = 0; r < 8; ++r)
      buf[j * VSTR + (m0 + r + 8 * hi)] = (__bf16)(acv[r] + vj);
  }
  __syncthreads();
  // cooperative coalesced store: thread t = output row d, 128 contiguous n (256B)
  {
    __bf16* dst = vals_t + (bb * DIM + t) * NTOK + nblk;
    const __bf16* src = buf + t * VSTR;
#pragma unroll
    for (int c = 0; c < 16; ++c)
      *(v8bf*)(dst + c * 8) = *(const v8bf*)(src + c * 8);
  }
}

// ---------------- q = Wq @ LN(slots) + bq, padded to 16 rows (rows 7..15 pre-zeroed).
// Also zeroes the per-slot attention normalizer for the upcoming k_attn1. ----
__global__ void __launch_bounds__(256)
k_qprep(const float* __restrict__ slots, const float* __restrict__ g,
        const float* __restrict__ be, const float* __restrict__ Wq,
        const float* __restrict__ bq, __bf16* __restrict__ qb,
        float* __restrict__ gsum) {
  __shared__ float s[DIM];
  __shared__ float red[8];
  const int t = threadIdx.x;
  const int b = blockIdx.x / NSLOTS, m = blockIdx.x % NSLOTS;
  if (m == 0 && t < 16) gsum[(long long)b * 16 + t] = 0.f;
  float v = slots[((long long)b * NSLOTS + m) * DIM + t];
  float mu = block_sum256(v, red) * (1.f / DIM);
  float dv = v - mu;
  float var = block_sum256(dv * dv, red) * (1.f / DIM);
  float rs = rsqrtf(var + LN_EPS);
  s[t] = dv * rs * g[t] + be[t];
  __syncthreads();
  const float4* w4 = (const float4*)(Wq + (size_t)t * DIM);
  const float4* s4 = (const float4*)s;
  float acc = bq[t];
  for (int i = 0; i < DIM / 4; ++i) {
    float4 wv = w4[i], sv = s4[i];
    acc += wv.x * sv.x + wv.y * sv.y + wv.z * sv.z + wv.w * sv.w;
  }
  qb[((long long)b * 16 + m) * DIM + t] = (__bf16)acc;
}

// ---------------- attention phase 1: dots (WMMA) + softmax over slots ----------------
// grid = BATCH*NSEG workgroups; each handles NTOK/NSEG tokens of one batch.
// Per-slot sums reduced in LDS, then one global f32 atomic per (WG, slot).
__global__ void __launch_bounds__(256)
k_attn1(const __bf16* __restrict__ qb, const __bf16* __restrict__ keys,
        __bf16* __restrict__ attn, float* __restrict__ gsum) {
  __shared__ float s_sum[16];
  const int t = threadIdx.x, wave = t >> 5, lane = t & 31, hi = lane >> 4, lo = lane & 15;
  const long long b = blockIdx.x / NSEG;
  const int seg = blockIdx.x % NSEG;
  if (t < 16) s_sum[t] = 0.f;
  __syncthreads();

  v16bf a[8];
  {
    const __bf16* qrow = qb + (b * 16 + lo) * DIM;
#pragma unroll
    for (int c = 0; c < 8; ++c) {
      int kA = c * 32 + hi * 16;
      a[c] = ld_pair(qrow + kA, qrow + kA + 8);
    }
  }

  float lsum[NSLOTS];
#pragma unroll
  for (int i = 0; i < NSLOTS; ++i) lsum[i] = 0.f;

  const int tile0 = seg * (256 / NSEG);
  for (int tile = tile0 + wave; tile < tile0 + 256 / NSEG; tile += 8) {
    const int n0 = tile * 16;
    v8f acc = {};
#pragma unroll
    for (int c = 0; c < 8; ++c) {
      const __bf16* p = keys + (b * NTOK + n0 + lo) * DIM + c * 32 + hi * 16;
      acc = wmma_bf16(a[c], ld_pair(p, p + 8), acc);
    }
    // rows 0..6 of column n0+lane live entirely in lanes 0..15 (C layout)
    if (lane < 16) {
      float v[NSLOTS], mx = -1e30f;
#pragma unroll
      for (int i = 0; i < NSLOTS; ++i) { v[i] = acc[i] * QSCALE; mx = fmaxf(mx, v[i]); }
      float s = 0.f;
#pragma unroll
      for (int i = 0; i < NSLOTS; ++i) { v[i] = __expf(v[i] - mx); s += v[i]; }
      float inv = 1.f / s;
      __bf16* ap = attn + (b * 16) * NTOK + n0 + lane;
#pragma unroll
      for (int i = 0; i < NSLOTS; ++i) {
        float aa = v[i] * inv + A_EPS;
        ap[(long long)i * NTOK] = (__bf16)aa;   // lanes contiguous over n: 32B/instr
        lsum[i] += aa;
      }
#pragma unroll
      for (int i = NSLOTS; i < 16; ++i) ap[(long long)i * NTOK] = (__bf16)0.f;
    }
  }
  if (lane < 16) {
#pragma unroll
    for (int i = 0; i < NSLOTS; ++i) atomicAdd(&s_sum[i], lsum[i]);
  }
  __syncthreads();
  if (t < NSLOTS) atomicAdd(&gsum[b * 16 + t], s_sum[t]);
}

// ---------------- attention phase 2: updates = attn @ values (WMMA) ----------------
// grid = BATCH*2 workgroups, one 16-wide d-tile per wave.
// out^T form: A = values_t rows (d), B = attn rows (m); renorm folded in as 1/sum[m].
__global__ void __launch_bounds__(256)
k_attn2(const __bf16* __restrict__ attn, const __bf16* __restrict__ vals_t,
        const float* __restrict__ gsum, float* __restrict__ upd) {
  const int t = threadIdx.x, wave = t >> 5, lane = t & 31, hi = lane >> 4, lo = lane & 15;
  const long long b = blockIdx.x >> 1;
  const int half = blockIdx.x & 1;
  const int d0 = (half * 8 + wave) * 16;

  v8f acc = {};
  for (int n0 = 0; n0 < NTOK; n0 += 32) {
    const __bf16* pa = vals_t + (b * DIM + d0 + lo) * NTOK + n0 + hi * 16;  // A: row d
    const __bf16* pb = attn   + (b * 16 + lo) * NTOK + n0 + hi * 16;      // B: col m
    acc = wmma_bf16(ld_pair(pa, pa + 8), ld_pair(pb, pb + 8), acc);
  }
  float s = gsum[b * 16 + lo];
  float inv = (s > 0.f) ? 1.f / s : 0.f;
  float4 u0 = make_float4(acc[0] * inv, acc[1] * inv, acc[2] * inv, acc[3] * inv);
  float4 u1 = make_float4(acc[4] * inv, acc[5] * inv, acc[6] * inv, acc[7] * inv);
  float* dst = upd + (b * 16 + lo) * DIM + d0 + 8 * hi;
  ((float4*)dst)[0] = u0;
  ((float4*)dst)[1] = u1;
}

// ---------------- GRU cell + LN + MLP residual (448 rows, tiny: fp32, float4 dots) ----
__global__ void __launch_bounds__(256)
k_gru_ff(const float* __restrict__ upd, const float* __restrict__ sin,
         const float* __restrict__ Wih, const float* __restrict__ bih,
         const float* __restrict__ Whh, const float* __restrict__ bhh,
         const float* __restrict__ lng, const float* __restrict__ lnb,
         const float* __restrict__ W1, const float* __restrict__ b1,
         const float* __restrict__ W2, const float* __restrict__ b2,
         float* __restrict__ sout) {
  __shared__ float xs[DIM], hs[DIM], h2[DIM], y1[DIM];
  __shared__ float red[8];
  const int t = threadIdx.x;
  const int b = blockIdx.x / NSLOTS, m = blockIdx.x % NSLOTS;
  const float x = upd[((long long)b * 16 + m) * DIM + t];
  const float h = sin[((long long)b * NSLOTS + m) * DIM + t];
  xs[t] = x; hs[t] = h;
  __syncthreads();

  float gi[3], gh[3];
#pragma unroll
  for (int gidx = 0; gidx < 3; ++gidx) {
    const float4* wi = (const float4*)(Wih + (size_t)(gidx * DIM + t) * DIM);
    const float4* wh = (const float4*)(Whh + (size_t)(gidx * DIM + t) * DIM);
    const float4* x4 = (const float4*)xs;
    const float4* h4 = (const float4*)hs;
    float ai = bih[gidx * DIM + t], ah = bhh[gidx * DIM + t];
    for (int i = 0; i < DIM / 4; ++i) {
      float4 wv = wi[i], xv = x4[i];
      ai += wv.x * xv.x + wv.y * xv.y + wv.z * xv.z + wv.w * xv.w;
      float4 hv = wh[i], hx = h4[i];
      ah += hv.x * hx.x + hv.y * hx.y + hv.z * hx.z + hv.w * hx.w;
    }
    gi[gidx] = ai; gh[gidx] = ah;
  }
  float r = 1.f / (1.f + __expf(-(gi[0] + gh[0])));
  float z = 1.f / (1.f + __expf(-(gi[1] + gh[1])));
  float nn = tanhf(gi[2] + r * gh[2]);
  float sn = (1.f - z) * nn + z * h;

  float mu = block_sum256(sn, red) * (1.f / DIM);
  float dv = sn - mu;
  float var = block_sum256(dv * dv, red) * (1.f / DIM);
  float rs = rsqrtf(var + LN_EPS);
  h2[t] = dv * rs * lng[t] + lnb[t];
  __syncthreads();
  {
    const float4* w = (const float4*)(W1 + (size_t)t * DIM);
    const float4* v = (const float4*)h2;
    float acc = b1[t];
    for (int i = 0; i < DIM / 4; ++i) {
      float4 wv = w[i], hv = v[i];
      acc += wv.x * hv.x + wv.y * hv.y + wv.z * hv.z + wv.w * hv.w;
    }
    y1[t] = fmaxf(acc, 0.f);
  }
  __syncthreads();
  {
    const float4* w = (const float4*)(W2 + (size_t)t * DIM);
    const float4* v = (const float4*)y1;
    float acc = b2[t];
    for (int i = 0; i < DIM / 4; ++i) {
      float4 wv = w[i], yv = v[i];
      acc += wv.x * yv.x + wv.y * yv.y + wv.z * yv.z + wv.w * yv.w;
    }
    sout[((long long)b * NSLOTS + m) * DIM + t] = sn + acc;
  }
}

// ---------------- launch ----------------
extern "C" void kernel_launch(void* const* d_in, const int* in_sizes, int n_in,
                              void* d_out, int out_size, void* d_ws, size_t ws_size,
                              hipStream_t stream) {
  const float* inputs  = (const float*)d_in[0];
  const float* noise   = (const float*)d_in[1];
  const float* smean   = (const float*)d_in[2];
  const float* slstd   = (const float*)d_in[3];
  const float* ln_in_g = (const float*)d_in[4];
  const float* ln_in_b = (const float*)d_in[5];
  const float* ln_sl_g = (const float*)d_in[6];
  const float* ln_sl_b = (const float*)d_in[7];
  const float* ln_ff_g = (const float*)d_in[8];
  const float* ln_ff_b = (const float*)d_in[9];
  const float* Wq  = (const float*)d_in[10];
  const float* bq  = (const float*)d_in[11];
  const float* Wk  = (const float*)d_in[12];
  const float* bk  = (const float*)d_in[13];
  const float* Wv  = (const float*)d_in[14];
  const float* bv  = (const float*)d_in[15];
  const float* Wih = (const float*)d_in[16];
  const float* bih = (const float*)d_in[17];
  const float* Whh = (const float*)d_in[18];
  const float* bhh = (const float*)d_in[19];
  const float* W1  = (const float*)d_in[20];
  const float* b1  = (const float*)d_in[21];
  const float* W2  = (const float*)d_in[22];
  const float* b2  = (const float*)d_in[23];

  char* ws = (char*)d_ws;
  size_t off = 0;
  __bf16* keys   = (__bf16*)(ws + off); off += (size_t)BATCH * NTOK * DIM * 2;   // 134 MB
  __bf16* vals_t = (__bf16*)(ws + off); off += (size_t)BATCH * NTOK * DIM * 2;   // 134 MB
  __bf16* qb     = (__bf16*)(ws + off); off += (size_t)BATCH * 16 * DIM * 2;
  __bf16* attnb  = (__bf16*)(ws + off); off += (size_t)BATCH * 16 * NTOK * 2;    // 8 MB
  float*  upd    = (float*) (ws + off); off += (size_t)BATCH * 16 * DIM * 4;
  float*  slots  = (float*) (ws + off); off += (size_t)BATCH * NSLOTS * DIM * 4;
  float*  gsum   = (float*) (ws + off); off += (size_t)BATCH * 16 * 4;
  __bf16* WkB    = (__bf16*)(ws + off); off += (size_t)DIM * DIM * 2;
  __bf16* WvB    = (__bf16*)(ws + off); off += (size_t)DIM * DIM * 2;

  const int nW = DIM * DIM;                 // 65536
  k_cvt_bf16<<<nW / 256, 256, 0, stream>>>(Wk, WkB, nW);
  k_cvt_bf16<<<nW / 256, 256, 0, stream>>>(Wv, WvB, nW);
  k_zero_bf16<<<(BATCH * 16 * DIM) / 256, 256, 0, stream>>>(qb, BATCH * 16 * DIM);
  k_slots_init<<<(BATCH * NSLOTS * DIM) / 256, 256, 0, stream>>>(
      smean, slstd, noise, slots, BATCH * NSLOTS * DIM);

  k_ln_kv<<<(BATCH * NTOK) / 128, 256, 0, stream>>>(
      inputs, ln_in_g, ln_in_b, WkB, WvB, bk, bv, keys, vals_t);

  for (int it = 0; it < 3; ++it) {
    k_qprep<<<BATCH * NSLOTS, 256, 0, stream>>>(slots, ln_sl_g, ln_sl_b, Wq, bq, qb, gsum);
    k_attn1<<<BATCH * NSEG, 256, 0, stream>>>(qb, keys, attnb, gsum);
    k_attn2<<<BATCH * 2, 256, 0, stream>>>(attnb, vals_t, gsum, upd);
    float* out = (it == 2) ? (float*)d_out : slots;
    k_gru_ff<<<BATCH * NSLOTS, 256, 0, stream>>>(
        upd, slots, Wih, bih, Whh, bhh, ln_ff_g, ln_ff_b, W1, b1, W2, b2, out);
  }
}
